// _TinyModel_34016140984517
// MI455X (gfx1250) — compile-verified, matrix operand order checked by reading, not run on hardware
//
#include <hip/hip_runtime.h>
#include <hip/hip_bf16.h>
#include <stdint.h>

// ---------------------------------------------------------------------------
// TinyModel: h = embed[ids]; scan: s = s@W^T + h_t; logits = hidden@lm_w^T + b
//   B=4, T=2048, D=1024, V=16000
// Phase 1: persistent-WG f32 WMMA recurrence (W slice resident in 320KB LDS)
// Phase 2: bf16 WMMA GEMM (async global->LDS staging) for the 268-GFLOP logits
//          matmul; NT stores keep lm_w resident in the 192MB L2.
// Workspace: bar(256B) | state 2x4x1024 f32 (32KB) | hidden bf16 (16MB) |
//            lm_w bf16 (32MB)   => ~48MB total
// ---------------------------------------------------------------------------

#define RB 4
#define RT 2048
#define RD 1024
#define RV 16000
#define NWG 16          // persistent workgroups in recurrence
#define WSTRIDE 1028    // padded LDS row stride (floats) to avoid bank conflicts

typedef __attribute__((ext_vector_type(2)))  float          v2f;
typedef __attribute__((ext_vector_type(4)))  float          v4f;
typedef __attribute__((ext_vector_type(8)))  float          v8f;
typedef __attribute__((ext_vector_type(4)))  unsigned int   v4u;
typedef __attribute__((ext_vector_type(4)))  unsigned short v4us;
typedef __attribute__((ext_vector_type(16))) __bf16         v16bf;

struct FragU { v4u lo, hi; };   // 32B, bit-castable to v16bf

__device__ __forceinline__ unsigned short f2bf(float f) {
  unsigned u = __builtin_bit_cast(unsigned, f);
  u += 0x7FFFu + ((u >> 16) & 1u);     // round-to-nearest-even
  return (unsigned short)(u >> 16);
}

// async 16B global -> LDS copy, tracked by ASYNCcnt (CDNA5 path)
__device__ __forceinline__ void async_cp16(unsigned lds_off, const void* gaddr) {
  asm volatile("global_load_async_to_lds_b128 %0, %1, off"
               :: "v"(lds_off), "v"(gaddr) : "memory");
}
__device__ __forceinline__ void async_drain() {
  asm volatile("s_wait_asynccnt 0x0" ::: "memory");
}

// ---------------------------------------------------------------------------
// Kernel 0: zero the global barrier + state buffer 0 (ws is poisoned 0xAA)
// ---------------------------------------------------------------------------
__global__ void init_ws(unsigned* __restrict__ bar, float* __restrict__ state0) {
  int i = threadIdx.x;
  if (i < 64) bar[i] = 0u;
  for (int j = i; j < RB * RD; j += 256) state0[j] = 0.0f;
}

// ---------------------------------------------------------------------------
// Kernel 1: lm_w f32 -> bf16 (read once -> NT load; bf16 result reused from L2)
// ---------------------------------------------------------------------------
__global__ __launch_bounds__(256)
void cvt_bf16(const float* __restrict__ in, unsigned short* __restrict__ out) {
  size_t i = (size_t)blockIdx.x * 256 + threadIdx.x;   // one float4 per thread
  v4f x = __builtin_nontemporal_load((const v4f*)(in + i * 4));
  v4us y;
  y.x = f2bf(x.x); y.y = f2bf(x.y); y.z = f2bf(x.z); y.w = f2bf(x.w);
  *(v4us*)(out + i * 4) = y;
}

// ---------------------------------------------------------------------------
// Kernel 2: recurrence. 16 persistent WGs x 256 threads (8 waves).
//   WG owns 64 output dims; its W rows live in LDS (64*1028*4 = 257KB).
//   State (padded to M=16; rows 4..15 are hard zeros fed from a zeroed LDS
//   pad with lane-stride 0 -> every lane does an unconditional ds_load_b64).
//   8 waves = 4 N-tiles x 2 K-halves; two interleaved accumulators break the
//   serial WMMA RAW chain; partial C tiles reduced through LDS.
//   Cross-WG sync: monotonic atomic counter + s_sleep spin.
// ---------------------------------------------------------------------------
__global__ __launch_bounds__(256)
void recurrence(const int* __restrict__ ids, const float* __restrict__ embed,
                const float* __restrict__ w_state, float* state,
                unsigned* bar, unsigned short* __restrict__ hidden) {
  extern __shared__ char smem[];
  float* Wlds    = (float*)smem;                                   // 64 x 1028
  float* Slocal  = (float*)(smem + 64 * WSTRIDE * 4);              // 4 x 1028
  float* partial = (float*)(smem + (64 + 4) * WSTRIDE * 4);        // 1024
  float* zeroPad = partial + 1024;                                 // 16

  const int tid   = threadIdx.x;
  const int lane  = tid & 31;
  const int wave  = tid >> 5;
  const int tile  = wave >> 1;      // 0..3 : which 16-dim N tile
  const int khalf = wave & 1;       // 0/1  : K range half
  const int dbase = blockIdx.x * 64;

  // Preload this WG's 64 rows of W into LDS (row-major, padded stride).
  for (int idx = tid; idx < 64 * 256; idx += 256) {
    int r  = idx >> 8;
    int cg = (idx & 255) << 2;
    v4f w = *(const v4f*)(w_state + (size_t)(dbase + r) * RD + cg);
    *(v4f*)(Wlds + r * WSTRIDE + cg) = w;
  }
  if (tid < 16) zeroPad[tid] = 0.0f;
  __syncthreads();

  const int m16 = lane & 15;
  const int khi = (lane >> 4) << 1;       // f32 A/B frag: K = khi + {0,1}
  const bool aOK = (m16 < RB);
  // Inactive (zero-row) lanes read the zero pad with stride 0 (LDS same-addr
  // broadcast) -> uniform unconditional ds_load_b64 for the A fragment.
  const float* ApBase = aOK ? (Slocal + m16 * WSTRIDE + khi) : zeroPad;
  const int astep = aOK ? 4 : 0;
  const int k0 = khalf * 512;
  int p = 0;

  for (int t = 0; t < RT; ++t) {
    const volatile float* Sin = state + p * (RB * RD);
    float* Sout = state + (p ^ 1) * (RB * RD);

    // Stage state rows 0..3 into LDS (volatile: written by other WGPs).
    for (int i = tid; i < RB * RD; i += 256)
      Slocal[(i >> 10) * WSTRIDE + (i & 1023)] = Sin[i];
    __syncthreads();

    // Two interleaved accumulators halve the WMMA->WMMA RAW chain depth.
    v8f c0 = {0, 0, 0, 0, 0, 0, 0, 0};
    v8f c1 = {0, 0, 0, 0, 0, 0, 0, 0};
    const float* Ap = ApBase + (aOK ? k0 : 0);
    const float* Bp = Wlds + (tile * 16 + m16) * WSTRIDE + khi + k0;
#pragma unroll 4
    for (int it = 0; it < 64; ++it) {
      v2f a0 = *(const v2f*)Ap; Ap += astep;
      v2f b0 = *(const v2f*)Bp; Bp += 4;
      c0 = __builtin_amdgcn_wmma_f32_16x16x4_f32(false, a0, false, b0,
                                                 (short)0, c0, false, false);
      v2f a1 = *(const v2f*)Ap; Ap += astep;
      v2f b1 = *(const v2f*)Bp; Bp += 4;
      c1 = __builtin_amdgcn_wmma_f32_16x16x4_f32(false, a1, false, b1,
                                                 (short)0, c1, false, false);
    }
    v8f c = c0 + c1;

    if (khalf == 1) {
      float* pp = partial + tile * 256 + lane * 8;
#pragma unroll
      for (int i = 0; i < 8; ++i) pp[i] = c[i];
    }
    __syncthreads();

    if (khalf == 0) {
      const float* pp = partial + tile * 256 + lane * 8;
#pragma unroll
      for (int i = 0; i < 8; ++i) c[i] += pp[i];
      // C layout: element v of lanes 0..15 is row m=v -> batches 0..3 live in
      // c[0..3] of lanes 0..15. Fuse embedding gather + bf16 hidden store.
      if (lane < 16) {
        int d = dbase + tile * 16 + lane;
#pragma unroll
        for (int b = 0; b < RB; ++b) {
          int id = ids[b * RT + t];
          float val = c[b] + embed[(size_t)id * RD + d];
          Sout[b * RD + d] = val;
          hidden[((size_t)b * RT + t) * RD + d] = f2bf(val);
        }
      }
    }
    __threadfence();
    __syncthreads();
    if (tid == 0) {
      __hip_atomic_fetch_add(bar, 1u, __ATOMIC_RELEASE, __HIP_MEMORY_SCOPE_AGENT);
      const unsigned tgt = (unsigned)(NWG * (t + 1));
      while (__hip_atomic_load(bar, __ATOMIC_ACQUIRE, __HIP_MEMORY_SCOPE_AGENT) < tgt)
        __builtin_amdgcn_s_sleep(2);
    }
    __syncthreads();
    p ^= 1;
  }
}

// ---------------------------------------------------------------------------
// Kernel 3: logits GEMM. C[8192x16000] = Hbf16 * lm_w_bf16^T + bias.
//   BM=BN=128, BK=32, 8 waves (4x2), 2x4 16x16 tiles per wave.
//   Double-buffered LDS staged via global_load_async_to_lds_b128 (ASYNCcnt),
//   v_wmma_f32_16x16x32_bf16 compute, nontemporal f32 stores for the 524MB
//   output so lm_w stays resident in L2.
// ---------------------------------------------------------------------------
__global__ __launch_bounds__(256)
void gemm_logits(const unsigned short* __restrict__ Abf,
                 const unsigned short* __restrict__ Bbf,
                 const float* __restrict__ bias, float* __restrict__ out) {
  __shared__ unsigned short As[2][128 * 40];   // padded stride 40 (80B)
  __shared__ unsigned short Bs[2][128 * 40];

  const int tid  = threadIdx.x;
  const int lane = tid & 31;
  const int wave = tid >> 5;
  const int wm = wave & 3;        // 4 waves along M
  const int wn = wave >> 2;       // 2 waves along N
  const int mbase = blockIdx.y * 128;
  const int nbase = blockIdx.x * 128;

  v8f zero = {0, 0, 0, 0, 0, 0, 0, 0};
  v8f acc[2][4];
#pragma unroll
  for (int mt = 0; mt < 2; ++mt)
#pragma unroll
    for (int nt = 0; nt < 4; ++nt) acc[mt][nt] = zero;

  auto stage = [&](int kc, int bufi) {
    int kb = kc * 32;
#pragma unroll
    for (int s = 0; s < 2; ++s) {
      int seg = tid + s * 256;          // 512 x 16B segments per operand
      int row = seg >> 2;
      int c8  = (seg & 3) << 3;
      async_cp16((unsigned)(uintptr_t)&As[bufi][row * 40 + c8],
                 (const void*)(Abf + (size_t)(mbase + row) * RD + kb + c8));
      async_cp16((unsigned)(uintptr_t)&Bs[bufi][row * 40 + c8],
                 (const void*)(Bbf + (size_t)(nbase + row) * RD + kb + c8));
    }
  };

  stage(0, 0);
  async_drain();
  __syncthreads();

  const int koff = (lane >> 4) << 3;    // lane half selects K {0-7,16-23} vs {8-15,24-31}
  for (int kc = 0; kc < 32; ++kc) {
    const int bufi = kc & 1;
    if (kc + 1 < 32) stage(kc + 1, bufi ^ 1);   // async copy overlaps compute

    v16bf af[2], bf[4];
#pragma unroll
    for (int mt = 0; mt < 2; ++mt) {
      const unsigned short* pA =
          &As[bufi][(wm * 32 + mt * 16 + (lane & 15)) * 40 + koff];
      FragU f; f.lo = *(const v4u*)pA; f.hi = *(const v4u*)(pA + 16);
      af[mt] = __builtin_bit_cast(v16bf, f);
    }
#pragma unroll
    for (int nt = 0; nt < 4; ++nt) {
      const unsigned short* pB =
          &Bs[bufi][(wn * 64 + nt * 16 + (lane & 15)) * 40 + koff];
      FragU f; f.lo = *(const v4u*)pB; f.hi = *(const v4u*)(pB + 16);
      bf[nt] = __builtin_bit_cast(v16bf, f);
    }
#pragma unroll
    for (int mt = 0; mt < 2; ++mt)
#pragma unroll
      for (int nt = 0; nt < 4; ++nt)
        acc[mt][nt] = __builtin_amdgcn_wmma_f32_16x16x32_bf16(
            false, af[mt], false, bf[nt], (short)0, acc[mt][nt], false, false);

    async_drain();      // next buffer fully in LDS before the barrier
    __syncthreads();
  }

  // Epilogue: D element v of lane l -> (m = v + 8*(l>>4), n = l&15). Add bias.
  // NT stores: logits are written once, never re-read -> don't evict lm_w.
#pragma unroll
  for (int mt = 0; mt < 2; ++mt) {
    int mrow = mbase + wm * 32 + mt * 16 + ((lane >> 4) << 3);
#pragma unroll
    for (int nt = 0; nt < 4; ++nt) {
      int n = nbase + wn * 64 + nt * 16 + (lane & 15);
      float bv = bias[n];
#pragma unroll
      for (int v = 0; v < 8; ++v)
        __builtin_nontemporal_store(acc[mt][nt][v] + bv,
                                    &out[(size_t)(mrow + v) * RV + n]);
    }
  }
}

// ---------------------------------------------------------------------------
extern "C" void kernel_launch(void* const* d_in, const int* in_sizes, int n_in,
                              void* d_out, int out_size, void* d_ws, size_t ws_size,
                              hipStream_t stream) {
  const int*   ids   = (const int*)d_in[0];
  const float* embed = (const float*)d_in[1];
  const float* wst   = (const float*)d_in[2];
  const float* lmw   = (const float*)d_in[3];
  const float* lmb   = (const float*)d_in[4];
  float* out = (float*)d_out;

  char* ws = (char*)d_ws;
  unsigned*       bar    = (unsigned*)ws;                       // 256 B
  float*          state  = (float*)(ws + 256);                  // 32 KB (2 bufs)
  unsigned short* hidden = (unsigned short*)(ws + 256 + 2 * RB * RD * 4);
  unsigned short* lmwbf  = (unsigned short*)(ws + 256 + 2 * RB * RD * 4 +
                                             (size_t)RB * RT * RD * 2);

  init_ws<<<1, 256, 0, stream>>>(bar, state);
  cvt_bf16<<<(RV * RD) / (256 * 4), 256, 0, stream>>>(lmw, lmwbf);

  size_t smem = (size_t)((64 + 4) * WSTRIDE + 1024 + 16) * sizeof(float); // ~283.8KB
  recurrence<<<NWG, 256, smem, stream>>>(ids, embed, wst, state, bar, hidden);

  gemm_logits<<<dim3(RV / 128, (RB * RT) / 128), 256, 0, stream>>>(hidden, lmwbf, lmb, out);
}
